// Trans_Local_SSG_33655363732084
// MI455X (gfx1250) — matrix operand order repaired and checked by reading, not compiled, not run on hardware
//
#include <hip/hip_runtime.h>
#include <hip/hip_bf16.h>

// ---------------------------------------------------------------------------
// Trans_Local_SSG for MI455X (gfx1250, wave32, WMMA 16x16x32 f16)
// B=8, N=4096, S=1024, K=32, IN_CH=64, D=128, HEADS=4, hd=32, r=0.2
// ---------------------------------------------------------------------------

#define DEV __device__ __forceinline__

typedef __attribute__((ext_vector_type(16))) _Float16 v16h;
typedef __attribute__((ext_vector_type(8)))  float    v8f;

// ----- WMMA fragment helpers (C/D layout per CDNA5 ISA 7.12.2; A/B use contiguous k, wave32) --------

// A operand, 16x32 f16, from row-major storage (stride in halfs).
// lane L: row m = L&15, half hi = L>>4.  elem e -> K = hi*16 + e
// (contiguous, matching the B loaders).
DEV v16h wmma_load_a(const _Float16* base, int stride, int m0, int k0) {
  int lane = threadIdx.x & 31;
  int m = lane & 15, hi = lane >> 4;
  const _Float16* p = base + (size_t)(m0 + m) * stride + k0;
  v16h a;
#pragma unroll
  for (int e = 0; e < 8; ++e) a[e] = p[hi * 16 + e];
#pragma unroll
  for (int e = 0; e < 8; ++e) a[8 + e] = p[hi * 16 + 8 + e];
  return a;
}

// B operand, 32x16 f16 (KxN), from B-TRANSPOSED storage bt[N][K] row-major:
// lane L: col n = L&15, hi = L>>4, elem e -> K = hi*16 + e  (contiguous).
DEV v16h wmma_load_bt(const _Float16* bt, int stride, int n0, int k0) {
  int lane = threadIdx.x & 31;
  int n = lane & 15, hi = lane >> 4;
  const _Float16* p = bt + (size_t)(n0 + n) * stride + k0 + hi * 16;
  v16h b;
#pragma unroll
  for (int e = 0; e < 16; ++e) b[e] = p[e];
  return b;
}

// B operand gathered from row-major M[K][N] storage (strided).
DEV v16h wmma_load_b_gather(const _Float16* mm, int stride, int n0, int k0) {
  int lane = threadIdx.x & 31;
  int n = lane & 15, hi = lane >> 4;
  v16h b;
#pragma unroll
  for (int e = 0; e < 16; ++e) b[e] = mm[(size_t)(k0 + hi * 16 + e) * stride + n0 + n];
  return b;
}

// C/D 16x16 f32: lane L: n = L&15, hi = L>>4; VGPR r -> row hi*8+r.
DEV v8f wmma_load_c(const float* cm, int stride, int m0, int n0) {
  int lane = threadIdx.x & 31;
  int n = lane & 15, hi = lane >> 4;
  v8f c;
#pragma unroll
  for (int r = 0; r < 8; ++r) c[r] = cm[(size_t)(m0 + hi * 8 + r) * stride + n0 + n];
  return c;
}

DEV void wmma_store_d_f32(float* om, int stride, int m0, int n0, v8f d) {
  int lane = threadIdx.x & 31;
  int n = lane & 15, hi = lane >> 4;
#pragma unroll
  for (int r = 0; r < 8; ++r) om[(size_t)(m0 + hi * 8 + r) * stride + n0 + n] = d[r];
}

DEV void wmma_store_d_f16(_Float16* om, int stride, int m0, int n0, v8f d) {
  int lane = threadIdx.x & 31;
  int n = lane & 15, hi = lane >> 4;
#pragma unroll
  for (int r = 0; r < 8; ++r)
    om[(size_t)(m0 + hi * 8 + r) * stride + n0 + n] = (_Float16)d[r];
}

DEV v8f wmma32(v16h a, v16h b, v8f c) {
  return __builtin_amdgcn_wmma_f32_16x16x32_f16(false, a, false, b, (short)0, c,
                                                false, false);
}

// ---------------------------------------------------------------------------
// Kernel 1: fold weights into f16, transposed for contiguous B-fragments.
//  wtT[128][96] : augmented conv+BN+pos weight, transposed.
//    row c of W~:  c<3 rel, 3..66 feat, 67..69 abs xyz, 70 const-1, 71..95 pad
//  wqT/wkT/wvT/woT[128][128] : W^T in f16.
// ---------------------------------------------------------------------------
__global__ void prep_weights_kernel(const float* Wc, const float* bc,
                                    const float* gamma, const float* beta,
                                    const float* mean, const float* var,
                                    const float* Wq, const float* Wk,
                                    const float* Wv, const float* Wo,
                                    const float* Wp, const float* bp,
                                    _Float16* wtT, _Float16* wqT, _Float16* wkT,
                                    _Float16* wvT, _Float16* woT) {
  int tid = threadIdx.x + blockIdx.x * blockDim.x;
  int nthreads = gridDim.x * blockDim.x;
  for (int i = tid; i < 128 * 96; i += nthreads) {
    int n = i / 96, c = i % 96;
    float g = gamma[n] * rsqrtf(var[n] + 1e-5f);
    float v;
    if (c < 67)       v = Wc[n * 67 + c] * g;               // conv * BN scale
    else if (c < 70)  v = Wp[(c - 67) * 128 + n];           // pos embed
    else if (c == 70) v = bc[n] * g + (beta[n] - mean[n] * g) + bp[n]; // biases
    else              v = 0.0f;                             // K padding
    wtT[i] = (_Float16)v;
  }
  for (int i = tid; i < 128 * 128; i += nthreads) {
    int n = i >> 7, k = i & 127;
    wqT[i] = (_Float16)Wq[k * 128 + n];
    wkT[i] = (_Float16)Wk[k * 128 + n];
    wvT[i] = (_Float16)Wv[k * 128 + n];
    woT[i] = (_Float16)Wo[k * 128 + n];
  }
}

// ---------------------------------------------------------------------------
// Kernel 2: farthest point sampling. One block per batch, 1024 threads
// (4 points/thread). Cloud + running distance in LDS. Argmax via wave32
// shuffle reduction (no barriers) + one cross-wave LDS step: 2 barriers
// per serial iteration instead of ~11.
// ---------------------------------------------------------------------------
DEV void argmax_reduce_wave(float& bv, int& bi) {
#pragma unroll
  for (int off = 16; off > 0; off >>= 1) {
    float ov = __shfl_down(bv, off, 32);
    int oi = __shfl_down(bi, off, 32);
    if (ov > bv || (ov == bv && oi < bi)) { bv = ov; bi = oi; }
  }
}

__global__ __launch_bounds__(1024) void fps_kernel(const float* xyz, int* fidx) {
  __shared__ float sx[4096], sy[4096], sz[4096], sd[4096];
  __shared__ float rv[32];
  __shared__ int ri[32];
  __shared__ int sBest;

  int b = blockIdx.x;
  int tid = threadIdx.x;
  int wave = tid >> 5, lane = tid & 31;
  const float* X = xyz + (size_t)b * 3 * 4096;
  for (int p = tid; p < 4096; p += 1024) {
    sx[p] = X[p];
    sy[p] = X[4096 + p];
    sz[p] = X[8192 + p];
    sd[p] = 1e10f;
  }
  if (tid == 0) sBest = 0;
  __syncthreads();

  int* out = fidx + b * 1024;
  for (int it = 0; it < 1024; ++it) {
    int cur = sBest;
    if (tid == 0) out[it] = cur;
    float cx = sx[cur], cy = sy[cur], cz = sz[cur];
    float bv = -1.0f;
    int bi = 0;
#pragma unroll
    for (int rr = 0; rr < 4; ++rr) {
      int p = tid + rr * 1024;  // ascending p => strict '>' keeps first max
      float dx = sx[p] - cx, dy = sy[p] - cy, dz = sz[p] - cz;
      float nd = fminf(sd[p], dx * dx + dy * dy + dz * dz);
      sd[p] = nd;
      if (nd > bv) { bv = nd; bi = p; }
    }
    argmax_reduce_wave(bv, bi);
    if (lane == 0) { rv[wave] = bv; ri[wave] = bi; }
    __syncthreads();
    if (wave == 0) {
      bv = rv[lane];
      bi = ri[lane];
      argmax_reduce_wave(bv, bi);
      if (lane == 0) sBest = bi;
    }
    __syncthreads();
  }
}

// ---------------------------------------------------------------------------
// Kernel 3: ball query, one wave32 per group. Ballot + prefix-popcount
// ordered compaction preserves the reference's ascending-index selection;
// pad with first hit. Also emits new_xyz [B,3,S].
// ---------------------------------------------------------------------------
__global__ __launch_bounds__(128) void ball_query_kernel(const float* xyz,
                                                         const int* fidx,
                                                         int* gidx,
                                                         float* out_newxyz) {
  int wave = threadIdx.x >> 5;
  int lane = threadIdx.x & 31;
  int g = blockIdx.x * 4 + wave;  // 2048 blocks * 4 waves = 8192 groups
  int b = g >> 10, s = g & 1023;
  const float* X = xyz + (size_t)b * 3 * 4096;
  int fi = fidx[g];
  float cx = X[fi], cy = X[4096 + fi], cz = X[8192 + fi];
  if (lane == 0) {
    out_newxyz[(size_t)(b * 3 + 0) * 1024 + s] = cx;
    out_newxyz[(size_t)(b * 3 + 1) * 1024 + s] = cy;
    out_newxyz[(size_t)(b * 3 + 2) * 1024 + s] = cz;
  }

  int* gi = gidx + (size_t)g * 32;
  int cnt = 0, first = -1;
  for (int base = 0; base < 4096 && cnt < 32; base += 32) {
    int n = base + lane;
    float dx = X[n] - cx, dy = X[4096 + n] - cy, dz = X[8192 + n] - cz;
    bool in = (dx * dx + dy * dy + dz * dz <= 0.04f);  // keep if !(d2 > r^2)
    unsigned mask = (unsigned)__ballot(in);
    if (in) {
      int pos = cnt + __popc(mask & ((1u << lane) - 1u));
      if (pos < 32) gi[pos] = n;
    }
    if (first < 0 && mask) first = base + __ffs(mask) - 1;
    cnt += __popc(mask);
  }
  cnt = min(cnt, 32);
  if (first < 0) first = 4095;  // degenerate: matches jax OOB clamp
  if (lane < 32 - cnt) gi[cnt + lane] = first;
}

// ---------------------------------------------------------------------------
// Kernel 4: per-group fused conv+BN+pos GEMM, QKV, 4-head attention, O-proj
// + residual, max-pool. One workgroup (4 waves) per group; wave w owns
// output columns [32w,32w+32) for projections and head w for attention.
// ---------------------------------------------------------------------------
__global__ __launch_bounds__(128) void group_attn_kernel(
    const float* xyz, const float* points, const int* fidx, const int* gidx,
    const _Float16* wtT, const _Float16* wqT, const _Float16* wkT,
    const _Float16* wvT, const _Float16* woT, float* out_pts) {
  __shared__ _Float16 sF[32 * 96];    // augmented features (f16)
  __shared__ float    sXf[32 * 128];  // x (f32, for residual)
  __shared__ _Float16 sXh[32 * 128];  // x (f16, WMMA A operand)
  __shared__ _Float16 sQ[32 * 128];
  __shared__ _Float16 sK[32 * 128];
  __shared__ _Float16 sV[32 * 128];
  __shared__ float    sS[4 * 32 * 32];  // scores; later reused as f32 out
  __shared__ _Float16 sA[4 * 32 * 32];  // softmax(attn) f16
  __shared__ _Float16 sO[32 * 128];     // attn @ v (f16)

  int tid = threadIdx.x;
  int wave = tid >> 5;
  int b = blockIdx.x >> 10, s = blockIdx.x & 1023;

  // ---- Phase 1: gather + build augmented feature matrix F[32][96] --------
  {
    const float* X = xyz + (size_t)b * 3 * 4096;
    const float* P = points + (size_t)b * 64 * 4096;
    int fi = fidx[blockIdx.x];
    float cx = X[fi], cy = X[4096 + fi], cz = X[8192 + fi];
    int t = tid >> 2, q4 = tid & 3;  // token, quarter of the 96 columns
    int idx = gidx[(size_t)blockIdx.x * 32 + t];
    float px = X[idx], py = X[4096 + idx], pz = X[8192 + idx];
    for (int c = q4 * 24; c < q4 * 24 + 24; ++c) {
      float v;
      if (c < 3)        v = (c == 0) ? px - cx : (c == 1) ? py - cy : pz - cz;
      else if (c < 67)  v = P[(size_t)(c - 3) * 4096 + idx];
      else if (c < 70)  v = (c == 67) ? px : (c == 68) ? py : pz;
      else if (c == 70) v = 1.0f;
      else              v = 0.0f;
      sF[t * 96 + c] = (_Float16)v;
    }
  }
  __syncthreads();

  // ---- Phase 2: x = F @ W~  (32x96x128; BN/bias/pos folded in) -----------
  for (int mt = 0; mt < 2; ++mt)
    for (int nh = 0; nh < 2; ++nh) {
      int n0 = wave * 32 + nh * 16;
      v8f acc = {};
#pragma unroll
      for (int ks = 0; ks < 3; ++ks)
        acc = wmma32(wmma_load_a(sF, 96, mt * 16, ks * 32),
                     wmma_load_bt(wtT, 96, n0, ks * 32), acc);
      wmma_store_d_f32(sXf, 128, mt * 16, n0, acc);
      wmma_store_d_f16(sXh, 128, mt * 16, n0, acc);
    }
  __syncthreads();

  // ---- Phase 3: q,k,v projections (32x128x128 each) ----------------------
  for (int mt = 0; mt < 2; ++mt)
    for (int nh = 0; nh < 2; ++nh) {
      int n0 = wave * 32 + nh * 16;
      v8f aq = {}, ak = {}, av = {};
#pragma unroll
      for (int ks = 0; ks < 4; ++ks) {
        v16h a = wmma_load_a(sXh, 128, mt * 16, ks * 32);
        aq = wmma32(a, wmma_load_bt(wqT, 128, n0, ks * 32), aq);
        ak = wmma32(a, wmma_load_bt(wkT, 128, n0, ks * 32), ak);
        av = wmma32(a, wmma_load_bt(wvT, 128, n0, ks * 32), av);
      }
      wmma_store_d_f16(sQ, 128, mt * 16, n0, aq);
      wmma_store_d_f16(sK, 128, mt * 16, n0, ak);
      wmma_store_d_f16(sV, 128, mt * 16, n0, av);
    }
  __syncthreads();

  // ---- Phase 4: scores = q_h @ k_h^T per head (head = wave) --------------
  {
    int h = wave;
    const _Float16* qh = sQ + 32 * h;
    const _Float16* kh = sK + 32 * h;  // bt-layout: kh[n][k] = k_h^T[k][n]
    for (int mt = 0; mt < 2; ++mt)
      for (int nt = 0; nt < 2; ++nt) {
        v8f acc = {};
        acc = wmma32(wmma_load_a(qh, 128, mt * 16, 0),
                     wmma_load_bt(kh, 128, nt * 16, 0), acc);
        wmma_store_d_f32(sS + h * 32 * 32, 32, mt * 16, nt * 16, acc);
      }
  }
  __syncthreads();

  // ---- Phase 5: softmax over k (scale = hd^-0.5), one thread per row -----
  {
    const float scale = 0.17677669529663687f;  // 1/sqrt(32)
    int h = tid >> 5, row = tid & 31;
    const float* sr = sS + (h * 32 + row) * 32;
    _Float16* ar = sA + (h * 32 + row) * 32;
    float mx = -1e30f;
    for (int j = 0; j < 32; ++j) mx = fmaxf(mx, sr[j] * scale);
    float sum = 0.0f;
    for (int j = 0; j < 32; ++j) {
      float ev = __expf(sr[j] * scale - mx);
      sum += ev;
      ar[j] = (_Float16)ev;
    }
    float inv = 1.0f / sum;
    for (int j = 0; j < 32; ++j) ar[j] = (_Float16)((float)ar[j] * inv);
  }
  __syncthreads();

  // ---- Phase 6: o_h = attn_h @ v_h ---------------------------------------
  {
    int h = wave;
    for (int mt = 0; mt < 2; ++mt)
      for (int nt = 0; nt < 2; ++nt) {
        v8f acc = {};
        acc = wmma32(wmma_load_a(sA + h * 32 * 32, 32, mt * 16, 0),
                     wmma_load_b_gather(sV, 128, 32 * h + nt * 16, 0), acc);
        wmma_store_d_f16(sO, 128, mt * 16, 32 * h + nt * 16, acc);
      }
  }
  __syncthreads();

  // ---- Phase 7: out = x + o @ Wo  (reuse sS as f32 out buffer) -----------
  float* sOut = sS;
  for (int mt = 0; mt < 2; ++mt)
    for (int nh = 0; nh < 2; ++nh) {
      int n0 = wave * 32 + nh * 16;
      v8f acc = wmma_load_c(sXf, 128, mt * 16, n0);  // residual as C
#pragma unroll
      for (int ks = 0; ks < 4; ++ks)
        acc = wmma32(wmma_load_a(sO, 128, mt * 16, ks * 32),
                     wmma_load_bt(woT, 128, n0, ks * 32), acc);
      wmma_store_d_f32(sOut, 128, mt * 16, n0, acc);
    }
  __syncthreads();

  // ---- Phase 8: max over 32 tokens, write new_points [B,128,S] -----------
  {
    int d = tid;
    float mx = sOut[d];
    for (int t = 1; t < 32; ++t) mx = fmaxf(mx, sOut[t * 128 + d]);
    out_pts[((size_t)b * 128 + d) * 1024 + s] = mx;
  }
}

// ---------------------------------------------------------------------------
extern "C" void kernel_launch(void* const* d_in, const int* in_sizes, int n_in,
                              void* d_out, int out_size, void* d_ws,
                              size_t ws_size, hipStream_t stream) {
  const float* xyz    = (const float*)d_in[0];   // [8,3,4096]
  const float* points = (const float*)d_in[1];   // [8,64,4096]
  const float* Wc     = (const float*)d_in[2];   // [128,67]
  const float* bc     = (const float*)d_in[3];
  const float* gamma  = (const float*)d_in[4];
  const float* beta   = (const float*)d_in[5];
  const float* mean   = (const float*)d_in[6];
  const float* var    = (const float*)d_in[7];
  const float* Wq     = (const float*)d_in[8];   // [128,128]
  const float* Wk     = (const float*)d_in[9];
  const float* Wv     = (const float*)d_in[10];
  const float* Wo     = (const float*)d_in[11];
  const float* Wp     = (const float*)d_in[12];  // [3,128]
  const float* bp     = (const float*)d_in[13];

  char* ws = (char*)d_ws;
  int* fidx       = (int*)(ws + 0);            // 8192 * 4        = 32 KB
  int* gidx       = (int*)(ws + 32768);        // 8192*32*4       = 1 MB
  _Float16* wtT   = (_Float16*)(ws + 1081344); // 128*96*2        = 24 KB
  _Float16* wqT   = (_Float16*)(ws + 1105920); // 128*128*2 each
  _Float16* wkT   = (_Float16*)(ws + 1138688);
  _Float16* wvT   = (_Float16*)(ws + 1171456);
  _Float16* woT   = (_Float16*)(ws + 1204224);

  float* out = (float*)d_out;  // [8,3,1024] new_xyz ++ [8,128,1024] new_points

  prep_weights_kernel<<<8, 256, 0, stream>>>(Wc, bc, gamma, beta, mean, var, Wq,
                                             Wk, Wv, Wo, Wp, bp, wtT, wqT, wkT,
                                             wvT, woT);
  fps_kernel<<<8, 1024, 0, stream>>>(xyz, fidx);
  ball_query_kernel<<<2048, 128, 0, stream>>>(xyz, fidx, gidx, out);
  group_attn_kernel<<<8192, 128, 0, stream>>>(xyz, points, fidx, gidx, wtT, wqT,
                                              wkT, wvT, woT, out + 8 * 3 * 1024);
}
